// ModernAttention_50422916055212
// MI455X (gfx1250) — matrix operand order repaired and checked by reading, not compiled, not run on hardware
//
#include <hip/hip_runtime.h>
#include <stdint.h>

// ---------------------------------------------------------------------------
// MI455X (gfx1250) GQA attention block, bf16 WMMA pipeline with TDM + async
// global->LDS staging.
// B=2, S=2048, DIM=4096, H=32, KV=8, HD=128, REP=4.
// ---------------------------------------------------------------------------

typedef __attribute__((ext_vector_type(16))) __bf16   v16bf;
typedef __attribute__((ext_vector_type(8)))  float    v8f;
typedef __attribute__((ext_vector_type(4)))  uint32_t u32x4;
typedef __attribute__((ext_vector_type(8)))  uint32_t u32x8;

#define B_   2
#define S_   2048
#define DIM_ 4096
#define H_   32
#define KV_  8
#define HD_  128

__device__ __forceinline__ uint16_t f2bf(float f) {
  uint32_t u = __float_as_uint(f);
  uint32_t r = u + 0x7FFFu + ((u >> 16) & 1u);   // round-to-nearest-even
  return (uint16_t)(r >> 16);
}
__device__ __forceinline__ float bf2f(uint16_t h) {
  return __uint_as_float(((uint32_t)h) << 16);
}

union FragB { v16bf v; uint32_t u[8]; };

// ---------------------------------------------------------------------------
// f32 -> bf16 conversion (packed pair stores)
// ---------------------------------------------------------------------------
__global__ void k_cvt_bf16(const float* __restrict__ in, uint32_t* __restrict__ out, int n2) {
  int i = blockIdx.x * blockDim.x + threadIdx.x;
  if (i >= n2) return;
  float a = in[2 * i], b = in[2 * i + 1];
  out[i] = (uint32_t)f2bf(a) | ((uint32_t)f2bf(b) << 16);
}

// ---------------------------------------------------------------------------
// RoPE in-place on bf16 [B,S,heads,HD]; cos/sin are f32 [S,HD]
// ---------------------------------------------------------------------------
__global__ void k_rope(uint16_t* __restrict__ qk, const float* __restrict__ cosb,
                       const float* __restrict__ sinb, int heads, int total) {
  int i = blockIdx.x * blockDim.x + threadIdx.x;
  if (i >= total) return;                 // total = B*S*heads*(HD/2)
  int d = i % (HD_ / 2);
  int t = i / (HD_ / 2);
  int h = t % heads; t /= heads;
  int s = t % S_;
  int b = t / S_;
  size_t base = (((size_t)(b * S_ + s)) * heads + h) * HD_;
  float q1 = bf2f(qk[base + d]);
  float q2 = bf2f(qk[base + d + HD_ / 2]);
  float c1 = cosb[s * HD_ + d],            s1 = sinb[s * HD_ + d];
  float c2 = cosb[s * HD_ + d + HD_ / 2],  s2 = sinb[s * HD_ + d + HD_ / 2];
  qk[base + d]            = f2bf(q1 * c1 - q2 * s1);
  qk[base + d + HD_ / 2]  = f2bf(q2 * c2 + q1 * s2);
}

// ---------------------------------------------------------------------------
// Tiled bf16 GEMM: C[M,N] = A[M,K] * Bm[K,N], f32 accumulate via WMMA.
// Block tile 128x128, K-step 32, 8 waves; wave = 16 rows x 128 cols.
// A tile staged by the Tensor Data Mover (one descriptor per K-step, wave 0),
// with TDM pad_enable producing the 40-element padded LDS rows directly.
// B tile staged manually (needs transpose). Epilogue: f32 (Cf) or bf16 (Cb).
// ---------------------------------------------------------------------------
#define GT_M 128
#define GT_N 128
#define GT_K 32
#define LDA  40   // padded LDS stride (bf16 elems): 64B data + 16B pad

__global__ __launch_bounds__(256)
void k_gemm_bf16(const uint16_t* __restrict__ A, const uint16_t* __restrict__ Bm,
                 int M, int N, int K, float* __restrict__ Cf, uint16_t* __restrict__ Cb) {
  __shared__ uint16_t As[GT_M * LDA];   // [row][k]
  __shared__ uint16_t Bs[GT_N * LDA];   // [col][k]  (transposed)
  const int tid  = threadIdx.x;
  const int lane = tid & 31, w = tid >> 5;
  const int half = lane >> 4, nl = lane & 15;
  const int m0 = blockIdx.y * GT_M, n0 = blockIdx.x * GT_N;

  v8f acc[8];
  const v8f vz = {0.f, 0.f, 0.f, 0.f, 0.f, 0.f, 0.f, 0.f};
#pragma unroll
  for (int i = 0; i < 8; i++) acc[i] = vz;

  const int bk = tid >> 3, bseg = (tid & 7) * 16;   // B staging role
  const int mrow = w * 16 + nl;
  const uint32_t ldsA = (uint32_t)(uintptr_t)(&As[0]);

  for (int k0 = 0; k0 < K; k0 += GT_K) {
    // ---- stage A tile via TDM: 128x32 tile, row stride K, LDS-padded ----
    if (w == 0) {
      uint64_t ga = (uint64_t)(uintptr_t)(A + (size_t)m0 * K + k0);
      u32x4 g0;
      g0[0] = 1u;                                    // count=1, no gather
      g0[1] = ldsA;                                  // lds_addr
      g0[2] = (uint32_t)ga;                          // global_addr[31:0]
      g0[3] = (uint32_t)(ga >> 32) | (2u << 30);     // global_addr[56:32] | type=2
      u32x8 g1;
      g1[0] = (1u << 16)        // data_size = 2 bytes
            | (1u << 20)        // pad_enable
            | (3u << 22)        // pad_interval: 16 DWORDs (64B)
            | (3u << 25);       // pad_amount: 4 DWORDs (16B)
      uint32_t td0 = (uint32_t)K;       // tensor_dim0
      uint32_t td1 = (uint32_t)M;       // tensor_dim1
      g1[1] = (td0 & 0xFFFFu) << 16;
      g1[2] = (td0 >> 16) | ((td1 & 0xFFFFu) << 16);
      g1[3] = (td1 >> 16) | ((uint32_t)GT_K << 16);  // tile_dim0 = 32
      g1[4] = (uint32_t)GT_M;                        // tile_dim1 = 128, tile_dim2 = 0
      g1[5] = (uint32_t)K;                           // tensor_dim0_stride[31:0]
      g1[6] = 0u;                                    // stride hi / dim1_stride lo
      g1[7] = 0u;
      asm volatile("tensor_load_to_lds %0, %1" :: "s"(g0), "s"(g1) : "memory");
      __builtin_amdgcn_s_wait_tensorcnt(0);
    }
    // ---- stage B tile (transposed to [n][k]) ----
    {
      const uint4* gb = (const uint4*)(Bm + (size_t)(k0 + bk) * N + n0 + bseg);
      uint4 b0 = gb[0], b1 = gb[1];
      uint16_t tmp[16];
      *(uint4*)&tmp[0] = b0; *(uint4*)&tmp[8] = b1;
#pragma unroll
      for (int j = 0; j < 16; j++) Bs[(bseg + j) * LDA + bk] = tmp[j];
    }
    if (k0 + GT_K < K)   // prefetch next B K-tile -> global_prefetch_b8
      __builtin_prefetch(Bm + (size_t)(k0 + GT_K + bk) * N + n0 + bseg);
    __syncthreads();

    // ---- compute: 1 A-frag, 8 B-frags, 8 WMMAs per wave ----
    FragB afr;
#pragma unroll
    for (int p = 0; p < 8; p++) {
      int kk = 2 * p + ((p & 4) ? 8 : 0) + 8 * half;      // A layout (16x32 bf16)
      afr.u[p] = *(const uint32_t*)&As[mrow * LDA + kk];
    }
#pragma unroll
    for (int nt = 0; nt < 8; nt++) {
      FragB bfr;
      const int ncol = nt * 16 + nl;
#pragma unroll
      for (int p = 0; p < 8; p++) {
        int kk = 2 * p + ((p & 4) ? 8 : 0) + 8 * half;      // B layout (32x16 bf16)
        bfr.u[p] = *(const uint32_t*)&Bs[ncol * LDA + kk];
      }
      acc[nt] = __builtin_amdgcn_wmma_f32_16x16x32_bf16(
          false, afr.v, false, bfr.v, (short)0, acc[nt], false, false);
    }
    __syncthreads();
  }

  // ---- epilogue: D layout VGPR i -> row i + 8*half, col = lane%16 ----
  const int rbase = m0 + w * 16;
#pragma unroll
  for (int nt = 0; nt < 8; nt++) {
#pragma unroll
    for (int i = 0; i < 8; i++) {
      size_t r = (size_t)(rbase + i + 8 * half);
      size_t c = (size_t)(n0 + nt * 16 + nl);
      float v = acc[nt][i];
      if (Cf) Cf[r * N + c] = v;
      else    Cb[r * N + c] = f2bf(v);
    }
  }
}

// ---------------------------------------------------------------------------
// Flash attention: block = (qblock of 128 rows, head, batch), 8 waves.
// Wave = 16 query rows x HD=128 output; online softmax.
// K tile staged with async global->LDS (ASYNCcnt); V staged manually
// (transposed). All 8 waves share the head-group's K/V tiles.
// ---------------------------------------------------------------------------
#define LDK 136
#define LDV 40
#define LDP 40

__global__ __launch_bounds__(256)
void k_attn(const uint16_t* __restrict__ Q, const uint16_t* __restrict__ Kb,
            const uint16_t* __restrict__ Vb, uint16_t* __restrict__ O) {
  __shared__ uint16_t Ks[32 * LDK];        // [key][dim]
  __shared__ uint16_t Vs[HD_ * LDV];       // [dim][key] (transposed)
  __shared__ uint16_t Ps[8 * 16 * LDP];    // per-wave P scratch [row][key]

  const int tid = threadIdx.x, lane = tid & 31, w = tid >> 5;
  const int half = lane >> 4, nl = lane & 15;
  const int qb = blockIdx.x, h = blockIdx.y, b = blockIdx.z;
  const int g = h / (H_ / KV_);
  const int qrow0 = qb * 128 + w * 16;
  const float scale = 0.08838834764831845f;  // 1/sqrt(128)

  // Q fragments: 4 chunks of 32 dims (A layout)
  FragB qf[4];
  {
    const uint16_t* qp = Q + (((size_t)(b * S_ + qrow0 + nl)) * H_ + h) * HD_;
#pragma unroll
    for (int c = 0; c < 4; c++)
#pragma unroll
      for (int p = 0; p < 8; p++) {
        int kk = 32 * c + 2 * p + ((p & 4) ? 8 : 0) + 8 * half;
        qf[c].u[p] = *(const uint32_t*)(qp + kk);
      }
  }

  float mi[8], li[8];
  v8f o[8];
  const v8f vz = {0.f, 0.f, 0.f, 0.f, 0.f, 0.f, 0.f, 0.f};
#pragma unroll
  for (int i = 0; i < 8; i++) { mi[i] = -1e30f; li[i] = 0.f; o[i] = vz; }

  const int kmax = (qb + 1) * 128;
  const int skey = tid >> 3, sseg = (tid & 7) * 16;

  for (int kb0 = 0; kb0 < kmax; kb0 += 32) {
    // ---- stage K [key][dim] via async global->LDS; V transposed manually ----
    {
      const uint16_t* kp = Kb + (((size_t)(b * S_ + kb0 + skey)) * KV_ + g) * HD_ + sseg;
      uint32_t ldsk = (uint32_t)(uintptr_t)&Ks[skey * LDK + sseg];
      asm volatile("global_load_async_to_lds_b128 %0, %1, off"
                   :: "v"(ldsk), "v"(kp) : "memory");
      asm volatile("global_load_async_to_lds_b128 %0, %1, off offset:16"
                   :: "v"(ldsk), "v"(kp) : "memory");

      const uint16_t* vp = Vb + (((size_t)(b * S_ + kb0 + skey)) * KV_ + g) * HD_ + sseg;
      uint4 v0v = ((const uint4*)vp)[0], v1v = ((const uint4*)vp)[1];
      uint16_t tmp[16];
      *(uint4*)&tmp[0] = v0v; *(uint4*)&tmp[8] = v1v;
#pragma unroll
      for (int j = 0; j < 16; j++) Vs[(sseg + j) * LDV + skey] = tmp[j];

      asm volatile("s_wait_asynccnt 0x0" ::: "memory");
    }
    __syncthreads();

    // ---- scores: two 16-key subtiles, K-dim 128 = 4 WMMAs each ----
    v8f sc[2];
#pragma unroll
    for (int kt = 0; kt < 2; kt++) {
      v8f s = vz;
#pragma unroll
      for (int c = 0; c < 4; c++) {
        FragB kf;
#pragma unroll
        for (int p = 0; p < 8; p++) {
          int kk = 32 * c + 2 * p + ((p & 4) ? 8 : 0) + 8 * half;
          kf.u[p] = *(const uint32_t*)&Ks[(kt * 16 + nl) * LDK + kk];
        }
        s = __builtin_amdgcn_wmma_f32_16x16x32_bf16(
            false, qf[c].v, false, kf.v, (short)0, s, false, false);
      }
      sc[kt] = s;
    }

    // ---- causal mask + online softmax (rows: i + 8*half; cols: lane%16) ----
    float alpha[8];
#pragma unroll
    for (int i = 0; i < 8; i++) {
      int qg = qrow0 + i + 8 * half;
      float a = sc[0][i] * scale;  if (kb0 + nl      > qg) a = -1e30f;
      float c = sc[1][i] * scale;  if (kb0 + 16 + nl > qg) c = -1e30f;
      float t = fmaxf(a, c);
#pragma unroll
      for (int off = 1; off < 16; off <<= 1) t = fmaxf(t, __shfl_xor(t, off, 32));
      float nm = fmaxf(mi[i], t);
      alpha[i] = __expf(mi[i] - nm);
      mi[i] = nm;
      float p0 = __expf(a - nm);
      float p1 = __expf(c - nm);
      float rs = p0 + p1;
#pragma unroll
      for (int off = 1; off < 16; off <<= 1) rs += __shfl_xor(rs, off, 32);
      li[i] = li[i] * alpha[i] + rs;
      // park P (bf16) in per-wave LDS scratch, [row][key0..31]
      Ps[(w * 16 + i + 8 * half) * LDP + nl]      = f2bf(p0);
      Ps[(w * 16 + i + 8 * half) * LDP + 16 + nl] = f2bf(p1);
    }
#pragma unroll
    for (int nt = 0; nt < 8; nt++)
#pragma unroll
      for (int i = 0; i < 8; i++) o[nt][i] *= alpha[i];

    // ---- P@V: A-frag (16x32) from scratch, 8 WMMAs across HD ----
    FragB pf;
#pragma unroll
    for (int p = 0; p < 8; p++) {
      int kk = 2 * p + ((p & 4) ? 8 : 0) + 8 * half;
      pf.u[p] = *(const uint32_t*)&Ps[(w * 16 + nl) * LDP + kk];
    }
#pragma unroll
    for (int nt = 0; nt < 8; nt++) {
      FragB vf;
#pragma unroll
      for (int p = 0; p < 8; p++) {
        int kk = 2 * p + ((p & 4) ? 8 : 0) + 8 * half;
        vf.u[p] = *(const uint32_t*)&Vs[(nt * 16 + nl) * LDV + kk];
      }
      o[nt] = __builtin_amdgcn_wmma_f32_16x16x32_bf16(
          false, pf.v, false, vf.v, (short)0, o[nt], false, false);
    }
    __syncthreads();
  }

  // ---- normalize + store bf16 [B,S,H*HD] ----
#pragma unroll
  for (int nt = 0; nt < 8; nt++)
#pragma unroll
    for (int i = 0; i < 8; i++) {
      int r = qrow0 + i + 8 * half;
      float v = o[nt][i] / li[i];
      O[((size_t)(b * S_ + r)) * (H_ * HD_) + h * HD_ + nt * 16 + nl] = f2bf(v);
    }
}

// ---------------------------------------------------------------------------
// Host launcher
// ---------------------------------------------------------------------------
extern "C" void kernel_launch(void* const* d_in, const int* in_sizes, int n_in,
                              void* d_out, int out_size, void* d_ws, size_t ws_size,
                              hipStream_t stream) {
  (void)in_sizes; (void)n_in; (void)out_size; (void)ws_size;
  const float* x    = (const float*)d_in[0];
  const float* cosb = (const float*)d_in[1];
  const float* sinb = (const float*)d_in[2];
  const float* Wq   = (const float*)d_in[3];
  const float* Wk   = (const float*)d_in[4];
  const float* Wv   = (const float*)d_in[5];
  const float* Wo   = (const float*)d_in[6];
  float* out = (float*)d_out;

  const size_t nX  = (size_t)B_ * S_ * DIM_;        // 16.78M
  const size_t nWq = (size_t)DIM_ * H_ * HD_;       // 16.78M
  const size_t nWk = (size_t)DIM_ * KV_ * HD_;      // 4.19M
  const size_t nQ  = (size_t)B_ * S_ * H_ * HD_;
  const size_t nK  = (size_t)B_ * S_ * KV_ * HD_;

  char* ws = (char*)d_ws;
  size_t off = 0;
  auto alloc = [&](size_t elems) {
    void* p = ws + off;
    off += (elems * sizeof(uint16_t) + 255) & ~(size_t)255;
    return (uint16_t*)p;
  };
  uint16_t* xb  = alloc(nX);
  uint16_t* Wqb = alloc(nWq);
  uint16_t* Wkb = alloc(nWk);
  uint16_t* Wvb = alloc(nWk);
  uint16_t* Wob = alloc(nWq);
  uint16_t* Qb  = alloc(nQ);
  uint16_t* Kbb = alloc(nK);
  uint16_t* Vbb = alloc(nK);
  uint16_t* Ab  = alloc(nQ);

  auto cvt = [&](const float* src, uint16_t* dst, size_t n) {
    int n2 = (int)(n / 2);
    k_cvt_bf16<<<(n2 + 255) / 256, 256, 0, stream>>>(src, (uint32_t*)dst, n2);
  };
  cvt(x,  xb,  nX);
  cvt(Wq, Wqb, nWq);
  cvt(Wk, Wkb, nWk);
  cvt(Wv, Wvb, nWk);
  cvt(Wo, Wob, nWq);

  const int M = B_ * S_;  // 4096
  // QKV projections (bf16 epilogue)
  k_gemm_bf16<<<dim3((H_ * HD_) / GT_N, M / GT_M), 256, 0, stream>>>(
      xb, Wqb, M, H_ * HD_, DIM_, nullptr, Qb);
  k_gemm_bf16<<<dim3((KV_ * HD_) / GT_N, M / GT_M), 256, 0, stream>>>(
      xb, Wkb, M, KV_ * HD_, DIM_, nullptr, Kbb);
  k_gemm_bf16<<<dim3((KV_ * HD_) / GT_N, M / GT_M), 256, 0, stream>>>(
      xb, Wvb, M, KV_ * HD_, DIM_, nullptr, Vbb);

  // RoPE on Q and K
  {
    int tq = B_ * S_ * H_ * (HD_ / 2);
    k_rope<<<(tq + 255) / 256, 256, 0, stream>>>(Qb, cosb, sinb, H_, tq);
    int tk = B_ * S_ * KV_ * (HD_ / 2);
    k_rope<<<(tk + 255) / 256, 256, 0, stream>>>(Kbb, cosb, sinb, KV_, tk);
  }

  // Flash attention
  k_attn<<<dim3(S_ / 128, H_, B_), 256, 0, stream>>>(Qb, Kbb, Vbb, Ab);

  // Output projection (f32 epilogue straight to d_out)
  k_gemm_bf16<<<dim3(DIM_ / GT_N, M / GT_M), 256, 0, stream>>>(
      Ab, Wob, M, DIM_, DIM_, out, nullptr);
}
